// AdditiveAttention_1099511627849
// MI455X (gfx1250) — matrix operand order repaired and checked by reading, not compiled, not run on hardware
//
#include <hip/hip_runtime.h>
#include <hip/hip_bf16.h>

typedef __attribute__((ext_vector_type(16))) __bf16 v16bf;
typedef __attribute__((ext_vector_type(8)))  __bf16 v8bf;
typedef __attribute__((ext_vector_type(8)))  float  v8f;

#define B_SZ 32
#define S_SZ 2048
#define D_SZ 1024

// ---------------------------------------------------------------------------
// Pack U (D x D fp32, row-major, K=row, N=col) into wave32 WMMA B-fragment
// order as bf16.  Fragment (kb, nt) covers K = kb*32..+31, N = nt*16..+15.
// B layout used here (32x16, contiguous k): lanes 0-15 hold K=0..15 of column N=lane,
// lanes 16-31 hold K=16..31.  Each lane stores 16 contiguous bf16 (32 B).
// ---------------------------------------------------------------------------
__global__ __launch_bounds__(256) void pack_u_kernel(const float* __restrict__ U,
                                                     __bf16* __restrict__ Up) {
    int gid  = blockIdx.x * 256 + threadIdx.x;   // 65536 lanes total
    int lane = gid & 31;
    int frag = gid >> 5;                         // kb*64 + nt
    int nt   = frag & 63;
    int kb   = frag >> 6;
    int lc   = lane & 15;
    int hi   = lane >> 4;
    int col  = nt * 16 + lc;
    int kbase = kb * 32 + hi * 16;
    __bf16* dst = Up + (size_t)gid * 16;
#pragma unroll
    for (int e = 0; e < 16; ++e)
        dst[e] = (__bf16)U[(size_t)(kbase + e) * D_SZ + col];
}

// ---------------------------------------------------------------------------
// wx[b][d] = sum_k vector[b][k] * W[k][d]   (tiny GEMV, fp32)
// ---------------------------------------------------------------------------
__global__ __launch_bounds__(256) void wx_kernel(const float* __restrict__ vec,
                                                 const float* __restrict__ W,
                                                 float* __restrict__ wx) {
    int d = blockIdx.x * 256 + threadIdx.x;
    int b = blockIdx.y;
    const float* vb = vec + (size_t)b * D_SZ;
    float acc = 0.f;
    for (int k = 0; k < D_SZ; ++k)
        acc = fmaf(vb[k], W[(size_t)k * D_SZ + d], acc);
    wx[(size_t)b * D_SZ + d] = acc;
}

// ---------------------------------------------------------------------------
// Fused: scores[b][s] = sum_d tanh( (matrix[b,s,:] @ U)[d] + wx[b][d] ) * v[d]
// Block = 512 threads (16 waves) handling 64 s-rows x all 1024 N columns.
// A tile (64 x 1024) is staged once into LDS as bf16 in A-fragment order
// (128 KB), shared by all 16 waves.  Wave w owns columns [w*64, w*64+64)
// -> acc[4 m-tiles][4 n-tiles] of v8f.  B fragments stream from the packed
// U in global memory (2 MB, L2-resident).
// ---------------------------------------------------------------------------
__global__ __launch_bounds__(512) void scores_kernel(const float* __restrict__ M,
                                                     const __bf16* __restrict__ Up,
                                                     const float* __restrict__ wx,
                                                     const float* __restrict__ vvec,
                                                     float* __restrict__ scores) {
    __shared__ __bf16 Atile[4 * 32 * 32 * 16];   // 65536 bf16 = 128 KB
    __shared__ float  sm[64];

    const int tid  = threadIdx.x;
    const int wave = tid >> 5;
    const int lane = tid & 31;
    const int lc   = lane & 15;
    const int hi   = lane >> 4;
    const int b    = blockIdx.y;
    const int s0   = blockIdx.x * 64;
    const int n0   = wave * 64;

    if (tid < 64) sm[tid] = 0.f;

    // --- Stage A: 64 rows x 1024 K fp32 -> bf16 fragments in LDS.
    // Fragment (mt, kb) base = ((mt*32+kb)*32 + lane)*16 bf16.
    // A layout: lane<16: M=lc, K 0..15; lane>=16: M=lc, K 16..31.
#pragma unroll
    for (int i = 0; i < 16; ++i) {
        int run = tid + 512 * i;        // 8192 runs of 8 elements
        int row = run >> 7;             // 0..63
        int rk  = (run & 127) * 8;      // 0..1016, multiple of 8
        const float* ap = M + ((size_t)b * S_SZ + (size_t)(s0 + row)) * D_SZ + rk;
        float4 x0 = *(const float4*)(ap + 0);
        float4 x1 = *(const float4*)(ap + 4);
        int mt = row >> 4;
        int rl = row & 15;
        int kb = rk >> 5;
        int kk = rk & 31;                       // 0, 8, 16, 24
        int lh = kk >> 4;                       // which lane half holds this run
        int eb = kk & 8;                        // element offset within fragment
        v8bf o;
        o[0] = (__bf16)x0.x; o[1] = (__bf16)x0.y; o[2] = (__bf16)x0.z; o[3] = (__bf16)x0.w;
        o[4] = (__bf16)x1.x; o[5] = (__bf16)x1.y; o[6] = (__bf16)x1.z; o[7] = (__bf16)x1.w;
        *(v8bf*)(Atile + (size_t)((mt * 32 + kb) * 32 + (lh * 16 + rl)) * 16 + eb) = o;
    }
    __syncthreads();

    v8f acc[4][4] = {};

    for (int kb = 0; kb < 32; ++kb) {
        v16bf af[4];
#pragma unroll
        for (int mt = 0; mt < 4; ++mt)
            af[mt] = *(const v16bf*)(Atile + (size_t)((mt * 32 + kb) * 32 + lane) * 16);

        const __bf16* bp = Up + ((size_t)(kb * 64 + (n0 >> 4)) * 32 + lane) * 16;
#pragma unroll
        for (int nt = 0; nt < 4; ++nt) {
            v16bf bf_ = *(const v16bf*)(bp + (size_t)nt * 32 * 16);
            acc[0][nt] = __builtin_amdgcn_wmma_f32_16x16x32_bf16(
                false, af[0], false, bf_, (short)0, acc[0][nt], false, false);
            acc[1][nt] = __builtin_amdgcn_wmma_f32_16x16x32_bf16(
                false, af[1], false, bf_, (short)0, acc[1][nt], false, false);
            acc[2][nt] = __builtin_amdgcn_wmma_f32_16x16x32_bf16(
                false, af[2], false, bf_, (short)0, acc[2][nt], false, false);
            acc[3][nt] = __builtin_amdgcn_wmma_f32_16x16x32_bf16(
                false, af[3], false, bf_, (short)0, acc[3][nt], false, false);
        }
    }

    // --- Epilogue: tanh(acc + wx) dot v, reduced per s-row.
    // C layout: VGPR r of acc holds (M = r + hi*8 within tile, N = lc + nt*16).
    float rowsum[4][8];
#pragma unroll
    for (int mt = 0; mt < 4; ++mt)
#pragma unroll
        for (int r = 0; r < 8; ++r) rowsum[mt][r] = 0.f;

#pragma unroll
    for (int nt = 0; nt < 4; ++nt) {
        const int col   = n0 + nt * 16 + lc;
        const float wxv = wx[(size_t)b * D_SZ + col];
        const float vv  = vvec[col];
#pragma unroll
        for (int mt = 0; mt < 4; ++mt)
#pragma unroll
            for (int r = 0; r < 8; ++r) {
                float t = tanhf(acc[mt][nt][r] + wxv);
                rowsum[mt][r] = fmaf(t, vv, rowsum[mt][r]);
            }
    }
    // cross-lane reduce within each 16-lane half (same row set per half)
#pragma unroll
    for (int mt = 0; mt < 4; ++mt)
#pragma unroll
        for (int r = 0; r < 8; ++r) {
            float v = rowsum[mt][r];
            v += __shfl_xor(v, 1, 32);
            v += __shfl_xor(v, 2, 32);
            v += __shfl_xor(v, 4, 32);
            v += __shfl_xor(v, 8, 32);
            rowsum[mt][r] = v;
        }
    if (lc == 0) {
#pragma unroll
        for (int mt = 0; mt < 4; ++mt)
#pragma unroll
            for (int r = 0; r < 8; ++r)
                atomicAdd(&sm[mt * 16 + hi * 8 + r], rowsum[mt][r]);
    }
    __syncthreads();
    if (tid < 64)
        scores[(size_t)b * S_SZ + s0 + tid] = sm[tid];
}

// ---------------------------------------------------------------------------
// Row softmax over S=2048
// ---------------------------------------------------------------------------
__global__ __launch_bounds__(256) void softmax_kernel(const float* __restrict__ scores,
                                                      float* __restrict__ out) {
    __shared__ float red[256];
    const int b = blockIdx.x, tid = threadIdx.x;
    const float* row = scores + (size_t)b * S_SZ;

    float lmax = -3.402823466e38f;
    for (int s = tid; s < S_SZ; s += 256) lmax = fmaxf(lmax, row[s]);
    red[tid] = lmax; __syncthreads();
    for (int off = 128; off > 0; off >>= 1) {
        if (tid < off) red[tid] = fmaxf(red[tid], red[tid + off]);
        __syncthreads();
    }
    const float m = red[0]; __syncthreads();

    float lsum = 0.f;
    for (int s = tid; s < S_SZ; s += 256) lsum += __expf(row[s] - m);
    red[tid] = lsum; __syncthreads();
    for (int off = 128; off > 0; off >>= 1) {
        if (tid < off) red[tid] += red[tid + off];
        __syncthreads();
    }
    const float inv = 1.0f / red[0];

    for (int s = tid; s < S_SZ; s += 256)
        out[(size_t)b * S_SZ + s] = __expf(row[s] - m) * inv;
}

// ---------------------------------------------------------------------------
extern "C" void kernel_launch(void* const* d_in, const int* in_sizes, int n_in,
                              void* d_out, int out_size, void* d_ws, size_t ws_size,
                              hipStream_t stream) {
    (void)in_sizes; (void)n_in; (void)out_size; (void)ws_size;
    const float* vec = (const float*)d_in[0];   // [B, D]
    const float* mat = (const float*)d_in[1];   // [B, S, D]
    const float* W   = (const float*)d_in[2];   // [D, D]
    const float* U   = (const float*)d_in[3];   // [D, D]
    const float* v   = (const float*)d_in[4];   // [D, 1]
    float* out = (float*)d_out;                 // [B, S]

    char* ws = (char*)d_ws;
    __bf16* Up    = (__bf16*)ws;                              // 2 MB packed U
    float*  wx    = (float*)(ws + (2u << 20));                // 128 KB
    float*  score = (float*)(ws + (2u << 20) + (128u << 10)); // 256 KB

    pack_u_kernel<<<256, 256, 0, stream>>>(U, Up);
    wx_kernel<<<dim3(D_SZ / 256, B_SZ), 256, 0, stream>>>(vec, W, wx);
    scores_kernel<<<dim3(S_SZ / 64, B_SZ), 512, 0, stream>>>(mat, Up, wx, v, score);
    softmax_kernel<<<B_SZ, 256, 0, stream>>>(score, out);
}
